// GNN_Predictor_59562606461317
// MI455X (gfx1250) — matrix operand order repaired and checked by reading, not compiled, not run on hardware
//
#include <hip/hip_runtime.h>

// ---------------------------------------------------------------------------
// GNN predictor for MI455X (gfx1250): bf16 WMMA GEMMs (f32 accumulate),
// L2-resident h table, f32 atomic scatter-add aggregation.
// ---------------------------------------------------------------------------

typedef __attribute__((ext_vector_type(16))) __bf16 v16bf;
typedef __attribute__((ext_vector_type(8)))  float  v8f;

#define WMMA_BF16(A, B, C) \
    __builtin_amdgcn_wmma_f32_16x16x32_bf16(false, (A), false, (B), (short)0, (C), false, false)

static constexpr int   NN_   = 280000;          // nodes
static constexpr int   EE_   = 1600000;         // edges
static constexpr int   BB_   = 40000;           // graphs
static constexpr int   HD_   = 64;
static constexpr int   XD_   = 6;

// workspace layout (byte offsets, all 256B aligned)
static constexpr size_t OFF_HA  = 0;                                   // bf16 h ping  N*64*2
static constexpr size_t OFF_HB  = OFF_HA + (size_t)NN_ * HD_ * 2;      // bf16 h pong
static constexpr size_t OFF_ACC = OFF_HB + (size_t)NN_ * HD_ * 2;      // f32 accumulator N*64*4
static constexpr size_t OFF_W   = OFF_ACC + (size_t)NN_ * HD_ * 4;     // bf16 weights

// element offsets inside bf16 weight region
static constexpr size_t U_W1T  = 0;              // 3 x 64x64
static constexpr size_t U_W2T  = U_W1T + 3 * 4096;
static constexpr size_t U_KWT  = U_W2T + 3 * 4096;   // 3 x 64x128 (kw transposed)
static constexpr size_t U_FC3T = U_KWT + 3 * 8192;   // 64x448
static constexpr size_t U_W0T  = U_FC3T + 64 * 448;  // 64x64
static constexpr size_t U_WHT  = U_W0T + 4096;       // 3 x 64x64

// --------------------------- helpers ---------------------------------------

__device__ __forceinline__ unsigned short f2bf(float f) {
    unsigned int u = __float_as_uint(f);
    u += 0x7FFFu + ((u >> 16) & 1u);            // round-to-nearest-even
    return (unsigned short)(u >> 16);
}
__device__ __forceinline__ float bf2f(unsigned short h) {
    return __uint_as_float(((unsigned int)h) << 16);
}

// A fragment: two 16-byte chunks (elements 0..7 and 8..15)
__device__ __forceinline__ v16bf load_frag2(const unsigned short* p0,
                                            const unsigned short* p1) {
    v16bf r;
    uint4* rp = reinterpret_cast<uint4*>(&r);
    rp[0] = *reinterpret_cast<const uint4*>(p0);
    rp[1] = *reinterpret_cast<const uint4*>(p1);
    return r;
}
// B fragment: one contiguous 32-byte chunk (16 bf16 of one column)
__device__ __forceinline__ v16bf load_frag32B(const unsigned short* p) {
    return load_frag2(p, p + 8);
}

__device__ __forceinline__ float lrelu(float v) { return v > 0.f ? v : 0.01f * v; }

// --------------------------- prep: weight transpose + bf16 ------------------

__global__ void k_w_transpose(const float* __restrict__ src,
                              unsigned short* __restrict__ dst,
                              int rows, int cols) {
    int idx = blockIdx.x * 256 + threadIdx.x;
    if (idx < rows * cols) {
        int r = idx / cols, c = idx % cols;
        dst[(size_t)c * rows + r] = f2bf(src[(size_t)r * cols + c]);
    }
}

// --------------------------- layer 0 (6 -> 64), scalar ----------------------

__global__ void k0_res(const float* __restrict__ x,
                       const float* __restrict__ rw1, const float* __restrict__ rb1,
                       const float* __restrict__ rw2, const float* __restrict__ rb2,
                       float* __restrict__ acc) {
    int tid = blockIdx.x * 256 + threadIdx.x;
    int node = tid >> 6, c = tid & 63;
    if (node >= NN_) return;
    float xv[XD_], t[XD_];
    #pragma unroll
    for (int i = 0; i < XD_; ++i) xv[i] = x[(size_t)node * XD_ + i];
    #pragma unroll
    for (int j = 0; j < XD_; ++j) {
        float s = rb1[j];
        #pragma unroll
        for (int i = 0; i < XD_; ++i) s += xv[i] * rw1[i * XD_ + j];
        t[j] = s;
    }
    float r = rb2[c];
    #pragma unroll
    for (int j = 0; j < XD_; ++j) r += t[j] * rw2[j * HD_ + c];
    acc[(size_t)node * HD_ + c] = r;
}

__global__ void k0_msg(const float* __restrict__ x,
                       const long long* __restrict__ ei,
                       const float* __restrict__ kw,
                       float* __restrict__ acc) {
    int tid = blockIdx.x * 256 + threadIdx.x;
    int e = tid >> 6, c = tid & 63;
    if (e >= EE_) return;
    size_t s = (size_t)ei[e];
    size_t d = (size_t)ei[(size_t)EE_ + e];
    float m = 0.f;
    #pragma unroll
    for (int i = 0; i < XD_; ++i) m += x[d * XD_ + i] * kw[i * HD_ + c];
    #pragma unroll
    for (int i = 0; i < XD_; ++i) m += x[s * XD_ + i] * kw[(XD_ + i) * HD_ + c];
    atomicAdd(&acc[d * HD_ + c], lrelu(m));
}

__global__ void k_fin(const float* __restrict__ acc, unsigned short* __restrict__ h) {
    int tid = blockIdx.x * 256 + threadIdx.x;
    if (tid < NN_ * HD_) h[tid] = f2bf(acc[tid]);
}

// ------------------- residual WMMA: acc = (h@w1+b1)@w2+b2 -------------------
// one wave per 16-node tile; 4 waves / block

__global__ __launch_bounds__(128)
void k_res_w(const unsigned short* __restrict__ h,
             const unsigned short* __restrict__ w1T, const float* __restrict__ b1,
             const unsigned short* __restrict__ w2T, const float* __restrict__ b2,
             float* __restrict__ acc) {
    __shared__ __align__(16) unsigned short sm[4][16][72];
    const int wave = threadIdx.x >> 5, lane = threadIdx.x & 31;
    const int m = lane & 15, g = lane >> 4;
    const int base = (blockIdx.x * 4 + wave) * 16;

    // A = h[base..base+15][0..63]   (16x32 bf16 A layout: chunks at K=8g, 16+8g)
    const unsigned short* row = h + (size_t)(base + m) * HD_;
    v16bf a[2];
    a[0] = load_frag2(row + 8 * g,      row + 16 + 8 * g);
    a[1] = load_frag2(row + 32 + 8 * g, row + 48 + 8 * g);

    v8f c[4];
    #pragma unroll
    for (int nt = 0; nt < 4; ++nt) {
        float bv = b1[nt * 16 + m];
        #pragma unroll
        for (int i = 0; i < 8; ++i) c[nt][i] = bv;
    }
    #pragma unroll
    for (int ks = 0; ks < 2; ++ks)
        #pragma unroll
        for (int nt = 0; nt < 4; ++nt) {
            v16bf b = load_frag2(w1T + (size_t)(nt * 16 + m) * 64 + 32 * ks + 8 * g, w1T + (size_t)(nt * 16 + m) * 64 + 32 * ks + 16 + 8 * g);
            c[nt] = WMMA_BF16(a[ks], b, c[nt]);
        }

    // bounce intermediate through per-wave LDS (C layout -> bf16 row-major)
    #pragma unroll
    for (int nt = 0; nt < 4; ++nt)
        #pragma unroll
        for (int i = 0; i < 8; ++i)
            sm[wave][i + 8 * g][nt * 16 + m] = f2bf(c[nt][i]);

    const unsigned short* lr = &sm[wave][m][0];
    a[0] = load_frag2(lr + 8 * g,      lr + 16 + 8 * g);
    a[1] = load_frag2(lr + 32 + 8 * g, lr + 48 + 8 * g);

    #pragma unroll
    for (int nt = 0; nt < 4; ++nt) {
        float bv = b2[nt * 16 + m];
        #pragma unroll
        for (int i = 0; i < 8; ++i) c[nt][i] = bv;
    }
    #pragma unroll
    for (int ks = 0; ks < 2; ++ks)
        #pragma unroll
        for (int nt = 0; nt < 4; ++nt) {
            v16bf b = load_frag2(w2T + (size_t)(nt * 16 + m) * 64 + 32 * ks + 8 * g, w2T + (size_t)(nt * 16 + m) * 64 + 32 * ks + 16 + 8 * g);
            c[nt] = WMMA_BF16(a[ks], b, c[nt]);
        }
    #pragma unroll
    for (int nt = 0; nt < 4; ++nt)
        #pragma unroll
        for (int i = 0; i < 8; ++i)
            acc[(size_t)(base + i + 8 * g) * HD_ + nt * 16 + m] = c[nt][i];
}

// --------- message WMMA: acc[dst] += lrelu(cat(h[dst],h[src]) @ kw) ---------
// one wave per 16-edge tile; A fragments gathered per lane from the
// L2-resident bf16 h table; 4 k-steps of 32 (K = 128 = 64 dst + 64 src feats)

__global__ __launch_bounds__(128)
void k_msg_w(const unsigned short* __restrict__ h,
             const long long* __restrict__ ei,
             const unsigned short* __restrict__ kwT,
             float* __restrict__ acc) {
    const int wave = threadIdx.x >> 5, lane = threadIdx.x & 31;
    const int m = lane & 15, g = lane >> 4;
    const int base = (blockIdx.x * 4 + wave) * 16;
    const int e = base + m;

    __builtin_prefetch(&ei[e + 2048], 0, 1);
    __builtin_prefetch(&ei[(size_t)EE_ + e + 2048], 0, 1);

    const size_t sN = (size_t)ei[e];
    const size_t dN = (size_t)ei[(size_t)EE_ + e];
    const unsigned short* drow = h + dN * HD_;
    const unsigned short* srow = h + sN * HD_;

    v16bf a[4];
    a[0] = load_frag2(drow + 8 * g,      drow + 16 + 8 * g);   // dst K 0..31
    a[1] = load_frag2(drow + 32 + 8 * g, drow + 48 + 8 * g);   // dst K 32..63
    a[2] = load_frag2(srow + 8 * g,      srow + 16 + 8 * g);   // src K 0..31
    a[3] = load_frag2(srow + 32 + 8 * g, srow + 48 + 8 * g);   // src K 32..63

    v8f c[4] = {};
    #pragma unroll
    for (int ks = 0; ks < 4; ++ks)
        #pragma unroll
        for (int nt = 0; nt < 4; ++nt) {
            v16bf b = load_frag2(kwT + (size_t)(nt * 16 + m) * 128 + 32 * ks + 8 * g, kwT + (size_t)(nt * 16 + m) * 128 + 32 * ks + 16 + 8 * g);
            c[nt] = WMMA_BF16(a[ks], b, c[nt]);
        }

    // leaky-relu + scatter-add (C layout: vgpr i -> row i+8g, col = lane%16)
    #pragma unroll
    for (int nt = 0; nt < 4; ++nt)
        #pragma unroll
        for (int i = 0; i < 8; ++i) {
            int M = i + 8 * g;
            size_t dd = (size_t)ei[(size_t)EE_ + base + M];
            atomicAdd(&acc[dd * HD_ + nt * 16 + m], lrelu(c[nt][i]));
        }
}

// ------------- readout: mu = g@fc3+b ; 4-layer MLP ; out = z@wl+bl ----------
// graph rows are contiguous (7 nodes x 64 = 448 bf16), one wave per 16 graphs

__global__ __launch_bounds__(128)
void k_readout(const unsigned short* __restrict__ h,
               const unsigned short* __restrict__ fc3T, const float* __restrict__ fc3b,
               const unsigned short* __restrict__ w0T,  const float* __restrict__ b0,
               const unsigned short* __restrict__ whT,
               const float* __restrict__ wl, const float* __restrict__ bl,
               float* __restrict__ out) {
    __shared__ __align__(16) unsigned short sm[4][16][72];
    const int wave = threadIdx.x >> 5, lane = threadIdx.x & 31;
    const int m = lane & 15, g = lane >> 4;
    const int base = (blockIdx.x * 4 + wave) * 16;

    // fc3: [16,448] @ [448,64] + b
    const unsigned short* arow = h + (size_t)(base + m) * 448;
    v8f c[4];
    #pragma unroll
    for (int nt = 0; nt < 4; ++nt) {
        float bv = fc3b[nt * 16 + m];
        #pragma unroll
        for (int i = 0; i < 8; ++i) c[nt][i] = bv;
    }
    for (int ks = 0; ks < 14; ++ks) {
        v16bf a = load_frag2(arow + 32 * ks + 8 * g, arow + 32 * ks + 16 + 8 * g);
        #pragma unroll
        for (int nt = 0; nt < 4; ++nt) {
            v16bf b = load_frag2(fc3T + (size_t)(nt * 16 + m) * 448 + 32 * ks + 8 * g, fc3T + (size_t)(nt * 16 + m) * 448 + 32 * ks + 16 + 8 * g);
            c[nt] = WMMA_BF16(a, b, c[nt]);
        }
    }
    // mu -> LDS (no activation)
    #pragma unroll
    for (int nt = 0; nt < 4; ++nt)
        #pragma unroll
        for (int i = 0; i < 8; ++i)
            sm[wave][i + 8 * g][nt * 16 + m] = f2bf(c[nt][i]);

    // 4 chained 64x64 GEMMs: w0 (+b0, lrelu), wh[0..2] (lrelu)
    for (int layer = 0; layer < 4; ++layer) {
        const unsigned short* W = (layer == 0) ? w0T : whT + (size_t)(layer - 1) * 4096;
        const unsigned short* lr = &sm[wave][m][0];
        v16bf a0 = load_frag2(lr + 8 * g,      lr + 16 + 8 * g);
        v16bf a1 = load_frag2(lr + 32 + 8 * g, lr + 48 + 8 * g);
        #pragma unroll
        for (int nt = 0; nt < 4; ++nt) {
            float bv = (layer == 0) ? b0[nt * 16 + m] : 0.f;
            #pragma unroll
            for (int i = 0; i < 8; ++i) c[nt][i] = bv;
        }
        #pragma unroll
        for (int nt = 0; nt < 4; ++nt) {
            v16bf b = load_frag2(W + (size_t)(nt * 16 + m) * 64 + 8 * g, W + (size_t)(nt * 16 + m) * 64 + 16 + 8 * g);
            c[nt] = WMMA_BF16(a0, b, c[nt]);
            b = load_frag2(W + (size_t)(nt * 16 + m) * 64 + 32 + 8 * g, W + (size_t)(nt * 16 + m) * 64 + 48 + 8 * g);
            c[nt] = WMMA_BF16(a1, b, c[nt]);
        }
        #pragma unroll
        for (int nt = 0; nt < 4; ++nt)
            #pragma unroll
            for (int i = 0; i < 8; ++i)
                sm[wave][i + 8 * g][nt * 16 + m] = f2bf(lrelu(c[nt][i]));
    }

    // final 64 -> 1 dot (f32 weights), lanes 0..15 only (after all WMMA)
    if (lane < 16) {
        float s = 0.f;
        #pragma unroll
        for (int k = 0; k < 64; ++k) s += bf2f(sm[wave][lane][k]) * wl[k];
        out[base + lane] = s + bl[0];
    }
}

// --------------------------- launch ----------------------------------------

extern "C" void kernel_launch(void* const* d_in, const int* in_sizes, int n_in,
                              void* d_out, int out_size, void* d_ws, size_t ws_size,
                              hipStream_t stream) {
    const float*     x        = (const float*)d_in[0];
    const long long* ei       = (const long long*)d_in[1];
    const float*     l0_rw1   = (const float*)d_in[2];
    const float*     l0_rb1   = (const float*)d_in[3];
    const float*     l0_rw2   = (const float*)d_in[4];
    const float*     l0_rb2   = (const float*)d_in[5];
    const float*     l0_kw    = (const float*)d_in[6];
    const float*     rw1s     = (const float*)d_in[7];
    const float*     rb1s     = (const float*)d_in[8];
    const float*     rw2s     = (const float*)d_in[9];
    const float*     rb2s     = (const float*)d_in[10];
    const float*     kws      = (const float*)d_in[11];
    const float*     fc3_w    = (const float*)d_in[12];
    const float*     fc3_b    = (const float*)d_in[13];
    const float*     mlp_w0   = (const float*)d_in[14];
    const float*     mlp_b0   = (const float*)d_in[15];
    const float*     mlp_wh   = (const float*)d_in[16];
    const float*     mlp_wl   = (const float*)d_in[17];
    const float*     mlp_bl   = (const float*)d_in[18];

    char* ws = (char*)d_ws;
    unsigned short* hA   = (unsigned short*)(ws + OFF_HA);
    unsigned short* hB   = (unsigned short*)(ws + OFF_HB);
    float*          acc  = (float*)(ws + OFF_ACC);
    unsigned short* wsb  = (unsigned short*)(ws + OFF_W);
    unsigned short* w1T  = wsb + U_W1T;
    unsigned short* w2T  = wsb + U_W2T;
    unsigned short* kwT  = wsb + U_KWT;
    unsigned short* fc3T = wsb + U_FC3T;
    unsigned short* w0T  = wsb + U_W0T;
    unsigned short* whT  = wsb + U_WHT;

    auto TT = [&](const float* s, unsigned short* d, int r, int c) {
        k_w_transpose<<<(r * c + 255) / 256, 256, 0, stream>>>(s, d, r, c);
    };
    for (int l = 0; l < 3; ++l) {
        TT(rw1s + (size_t)l * 4096, w1T + (size_t)l * 4096, 64, 64);
        TT(rw2s + (size_t)l * 4096, w2T + (size_t)l * 4096, 64, 64);
        TT(kws  + (size_t)l * 8192, kwT + (size_t)l * 8192, 128, 64);
        TT(mlp_wh + (size_t)l * 4096, whT + (size_t)l * 4096, 64, 64);
    }
    TT(fc3_w, fc3T, 448, 64);
    TT(mlp_w0, w0T, 64, 64);

    // layer 0 (scalar): acc = res(x); acc += scatter(lrelu(msg)); hA = bf16(acc)
    k0_res<<<(NN_ * HD_) / 256 + 1, 256, 0, stream>>>(x, l0_rw1, l0_rb1, l0_rw2, l0_rb2, acc);
    k0_msg<<<(EE_ * HD_) / 256, 256, 0, stream>>>(x, ei, l0_kw, acc);
    k_fin<<<(NN_ * HD_) / 256, 256, 0, stream>>>(acc, hA);

    // layers 1..3 (WMMA)
    unsigned short* hin = hA;
    unsigned short* hout = hB;
    for (int l = 0; l < 3; ++l) {
        k_res_w<<<NN_ / 16 / 4, 128, 0, stream>>>(
            hin, w1T + (size_t)l * 4096, rb1s + (size_t)l * 64,
                 w2T + (size_t)l * 4096, rb2s + (size_t)l * 64, acc);
        k_msg_w<<<EE_ / 16 / 4, 128, 0, stream>>>(
            hin, ei, kwT + (size_t)l * 8192, acc);
        k_fin<<<(NN_ * HD_) / 256, 256, 0, stream>>>(acc, hout);
        unsigned short* t = hin; hin = hout; hout = t;
    }

    // readout + MLP head (WMMA)
    k_readout<<<BB_ / 16 / 4, 128, 0, stream>>>(
        hin, fc3T, fc3_b, w0T, mlp_b0, whT, mlp_wl, mlp_bl, (float*)d_out);
}
